// EquivNoiseGCN_61512521613945
// MI455X (gfx1250) — matrix operand order repaired and checked by reading, not compiled, not run on hardware
//
#include <hip/hip_runtime.h>
#include <hip/hip_bf16.h>
#include <math.h>

// ---------------- problem constants ----------------
#define NN 40000
#define EE 640000
#define INF_ 128
#define HH 256
#define CC 3
#define LL 128
#define MM 42              // L/3
#define NEGV (-42.0f/86.0f)

typedef __attribute__((ext_vector_type(16))) __bf16 v16bf;
typedef __attribute__((ext_vector_type(8)))  float  v8f;

__device__ __forceinline__ float silu_f(float x) { return x / (1.0f + __expf(-x)); }

// =====================================================================
// Weight pre-pack: B (K x 256) fp32 row-major -> bf16 fragments laid out so
// each lane's 16-element B-fragment is one contiguous 32-byte chunk:
//   Bp[((nt*KC + kc)*32 + lane)*16 + e] = bf16( B[(kc*32 + (lane>>4)*8 + (e&7) + (e>>3)*16)*256 + nt*16 + (lane&15)] )
// =====================================================================
__global__ void pack_b_kernel(const float* __restrict__ B, __bf16* __restrict__ Bp, int K) {
    const int KC = K / 32;
    const int total = 16 * KC * 32 * 16;
    const int idx = blockIdx.x * blockDim.x + threadIdx.x;
    if (idx >= total) return;
    const int e    = idx & 15;
    const int lane = (idx >> 4) & 31;
    const int rest = idx >> 9;
    const int kc   = rest % KC;
    const int nt   = rest / KC;
    const int k = kc * 32 + (lane >> 4) * 8 + (e & 7) + (e >> 3) * 16;
    const int n = nt * 16 + (lane & 15);
    Bp[idx] = (__bf16)B[(size_t)k * HH + n];
}

// =====================================================================
// WMMA GEMM: out(Nrows x 256) = epi( A(Nrows x K) @ B + bias )
// A fp32 row-major (cvt to bf16 on the fly, float4 loads); Bp pre-packed bf16.
// EPI: 0 = +bias, 1 = silu(+bias), 2 = +bias + residual
// One wave per 16-row tile; 4 waves/block; A frags cached in VGPRs.
// =====================================================================
template<int K, int EPI>
__global__ void wmma_gemm_kernel(const float* __restrict__ A,
                                 const __bf16* __restrict__ Bp,
                                 const float* __restrict__ bias,
                                 const float* __restrict__ res,
                                 float* __restrict__ out,
                                 int Nrows)
{
    constexpr int KC = K / 32;
    const int wave = threadIdx.x >> 5;
    const int lane = threadIdx.x & 31;
    const int row0 = (blockIdx.x * 4 + wave) * 16;
    if (row0 >= Nrows) return;                 // wave-uniform; EXEC all-1s inside
    const int m  = lane & 15;
    const int hi = lane >> 4;

    // ---- load & convert A fragments: 4x float4 per 16x32 fragment ----
    v16bf afrag[KC];
    const float* Arow = A + (size_t)(row0 + m) * K;
#pragma unroll
    for (int kc = 0; kc < KC; ++kc) {
        const float* base = Arow + kc * 32 + hi * 8;
        const float4 a0 = *(const float4*)(base);
        const float4 a1 = *(const float4*)(base + 4);
        const float4 a2 = *(const float4*)(base + 16);
        const float4 a3 = *(const float4*)(base + 20);
        afrag[kc][0]  = (__bf16)a0.x;  afrag[kc][1]  = (__bf16)a0.y;
        afrag[kc][2]  = (__bf16)a0.z;  afrag[kc][3]  = (__bf16)a0.w;
        afrag[kc][4]  = (__bf16)a1.x;  afrag[kc][5]  = (__bf16)a1.y;
        afrag[kc][6]  = (__bf16)a1.z;  afrag[kc][7]  = (__bf16)a1.w;
        afrag[kc][8]  = (__bf16)a2.x;  afrag[kc][9]  = (__bf16)a2.y;
        afrag[kc][10] = (__bf16)a2.z;  afrag[kc][11] = (__bf16)a2.w;
        afrag[kc][12] = (__bf16)a3.x;  afrag[kc][13] = (__bf16)a3.y;
        afrag[kc][14] = (__bf16)a3.z;  afrag[kc][15] = (__bf16)a3.w;
    }

    // ---- sweep 16 N-tiles of the 256-wide output ----
#pragma unroll 1
    for (int nt = 0; nt < 16; ++nt) {
        v8f acc = {};
        const int n = nt * 16 + m;
        const __bf16* bptr = Bp + (((size_t)nt * KC) * 32 + lane) * 16;
#pragma unroll
        for (int kc = 0; kc < KC; ++kc) {
            const v16bf bfrag = *(const v16bf*)(bptr + (size_t)kc * 512);
            acc = __builtin_amdgcn_wmma_f32_16x16x32_bf16(
                      false, afrag[kc], false, bfrag, (short)0, acc, false, false);
        }
        const float bv = bias[n];
#pragma unroll
        for (int r = 0; r < 8; ++r) {
            const int mr = r + hi * 8;                      // C/D layout
            float val = acc[r] + bv;
            if (EPI == 1) val = silu_f(val);
            if (EPI == 2) val += res[(size_t)(row0 + mr) * HH + n];
            out[(size_t)(row0 + mr) * HH + n] = val;
        }
    }
}

// =====================================================================
// Elementwise / reduction helpers
// =====================================================================
__global__ void zero_kernel(float* __restrict__ p, int n) {
    int i = blockIdx.x * blockDim.x + threadIdx.x;
    if (i < n) p[i] = 0.0f;
}

__global__ void copy_kernel(const float* __restrict__ s, float* __restrict__ d, int n) {
    int i = blockIdx.x * blockDim.x + threadIdx.x;
    if (i < n) d[i] = s[i];
}

__global__ void deg_kernel(const float* __restrict__ vals, const int* __restrict__ row,
                           float* __restrict__ deg) {
    int e = blockIdx.x * blockDim.x + threadIdx.x;
    if (e < EE) atomicAdd(&deg[row[e]], vals[e]);
}

__global__ void dn_kernel(float* __restrict__ deg) {     // in-place deg -> rsqrt(max(deg,1))
    int i = blockIdx.x * blockDim.x + threadIdx.x;
    if (i < NN) deg[i] = rsqrtf(fmaxf(deg[i], 1.0f));
}

__global__ void ev_kernel(const float* __restrict__ vals, const int* __restrict__ row,
                          const int* __restrict__ col, const float* __restrict__ dn,
                          float* __restrict__ ev) {
    int e = blockIdx.x * blockDim.x + threadIdx.x;
    if (e < EE) ev[e] = vals[e] * dn[col[e]] * dn[row[e]];
}

// nzt: 1.0 iff (# strictly greater) < M  (matches prob >= border incl. ties)
__global__ void nzt_kernel(const float* __restrict__ prob, float* __restrict__ z) {
    __shared__ float p[LL];
    const int n = blockIdx.x, t = threadIdx.x;
    const float mine = prob[(size_t)n * LL + t];
    p[t] = mine;
    __syncthreads();
    int cg = 0;
#pragma unroll 16
    for (int j = 0; j < LL; ++j) cg += (p[j] > mine);
    z[(size_t)n * LL + t] = (cg < MM) ? 1.0f : NEGV;
}

// LayerNorm over H=256, one wave per row
__global__ void layernorm_kernel(const float* __restrict__ x, float* __restrict__ y) {
    const int wave = threadIdx.x >> 5, lane = threadIdx.x & 31;
    const int rowi = blockIdx.x * 4 + wave;
    if (rowi >= NN) return;
    const float* xr = x + (size_t)rowi * HH;
    float v[8]; float s = 0.0f;
#pragma unroll
    for (int i = 0; i < 8; ++i) { v[i] = xr[lane + 32 * i]; s += v[i]; }
#pragma unroll
    for (int o = 16; o; o >>= 1) s += __shfl_xor(s, o, 32);
    const float mean = s * (1.0f / HH);
    float q = 0.0f;
#pragma unroll
    for (int i = 0; i < 8; ++i) { const float d = v[i] - mean; q += d * d; }
#pragma unroll
    for (int o = 16; o; o >>= 1) q += __shfl_xor(q, o, 32);
    const float inv = rsqrtf(q * (1.0f / HH) + 1e-5f);
    float* yr = y + (size_t)rowi * HH;
#pragma unroll
    for (int i = 0; i < 8; ++i) yr[lane + 32 * i] = (v[i] - mean) * inv;
}

// coeffZ = coeff * (x1 @ Wc + bc), Wc (256x3); one wave per node
__global__ void coeffz_kernel(const float* __restrict__ x1, const float* __restrict__ Wc,
                              const float* __restrict__ bc, float* __restrict__ cz) {
    const int wave = threadIdx.x >> 5, lane = threadIdx.x & 31;
    const int n = blockIdx.x * 4 + wave;
    if (n >= NN) return;
    const float* xr = x1 + (size_t)n * HH;
    float a0 = 0, a1 = 0, a2 = 0;
    for (int i = lane; i < HH; i += 32) {
        const float xv = xr[i];
        a0 += xv * Wc[i * 3 + 0];
        a1 += xv * Wc[i * 3 + 1];
        a2 += xv * Wc[i * 3 + 2];
    }
#pragma unroll
    for (int o = 16; o; o >>= 1) {
        a0 += __shfl_xor(a0, o, 32);
        a1 += __shfl_xor(a1, o, 32);
        a2 += __shfl_xor(a2, o, 32);
    }
    if (lane == 0) {
        const float coeff = rsqrtf((float)(CC + HH));
        cz[n * 3 + 0] = coeff * (a0 + bc[0]);
        cz[n * 3 + 1] = coeff * (a1 + bc[1]);
        cz[n * 3 + 2] = coeff * (a2 + bc[2]);
    }
}

// Zfull[n,o,l] = Z1[n,l] + bv[o] + sum_c Wv[o,c]*silu(cz[n,c]*Z1[n,l])
__global__ void zupdate_kernel(const float* __restrict__ Z1, const float* __restrict__ cz,
                               const float* __restrict__ Wv, const float* __restrict__ bv,
                               float* __restrict__ Zfull) {
    const int idx = blockIdx.x * blockDim.x + threadIdx.x;
    if (idx >= NN * LL) return;
    const int n = idx >> 7, l = idx & (LL - 1);
    const float z = Z1[idx];
    float s[CC];
#pragma unroll
    for (int c = 0; c < CC; ++c) s[c] = silu_f(cz[n * CC + c] * z);
#pragma unroll
    for (int o = 0; o < CC; ++o) {
        float val = bv[o];
#pragma unroll
        for (int c = 0; c < CC; ++c) val += Wv[o * CC + c] * s[c];
        Zfull[((size_t)n * CC + o) * LL + l] = z + val;
    }
}

// Z2 = silu(conv1x1(Wz,bz, vnorm(Zfull)))
__global__ void vnorm_kernel(const float* __restrict__ Zfull, const float* __restrict__ Wz,
                             const float* __restrict__ bz, float* __restrict__ Z2) {
    const int idx = blockIdx.x * blockDim.x + threadIdx.x;
    if (idx >= NN * LL) return;
    const int n = idx >> 7, l = idx & (LL - 1);
    float z[CC];
#pragma unroll
    for (int c = 0; c < CC; ++c) z[c] = Zfull[((size_t)n * CC + c) * LL + l];
    const float mean = (z[0] + z[1] + z[2]) * (1.0f / CC);
    float zc[CC], ss = 0.0f;
#pragma unroll
    for (int c = 0; c < CC; ++c) { zc[c] = z[c] - mean; ss += zc[c] * zc[c]; }
    const float std_ = sqrtf(ss * (1.0f / (CC - 1)));     // unbiased
    const float inv = 1.0f / (std_ + 1e-4f);
#pragma unroll
    for (int o = 0; o < CC; ++o) {
        float val = bz[o];
#pragma unroll
        for (int c = 0; c < CC; ++c) val += Wz[o * CC + c] * (zc[c] * inv);
        Z2[((size_t)n * CC + o) * LL + l] = silu_f(val);
    }
}

// edge-parallel SpMM scatter: out[row[e],f] += ev[e]*X[col[e],f]; block = F threads, 1 edge/block
__global__ void spmm_atomic_kernel(const float* __restrict__ X, const int* __restrict__ row,
                                   const int* __restrict__ col, const float* __restrict__ ev,
                                   float* __restrict__ out, int F) {
    const int e = blockIdx.x, f = threadIdx.x;
    const float w = ev[e];
    const int r = row[e], c = col[e];
    atomicAdd(&out[(size_t)r * F + f], w * X[(size_t)c * F + f]);
}

// =====================================================================
extern "C" void kernel_launch(void* const* d_in, const int* in_sizes, int n_in,
                              void* d_out, int out_size, void* d_ws, size_t ws_size,
                              hipStream_t stream) {
    const float* x     = (const float*)d_in[0];
    const int*   row   = (const int*)  d_in[1];
    const int*   col   = (const int*)  d_in[2];
    const float* vals  = (const float*)d_in[3];
    const float* prob  = (const float*)d_in[4];
    const float* W_emb = (const float*)d_in[5];
    const float* b_emb = (const float*)d_in[6];
    const float* W2a   = (const float*)d_in[7];
    const float* b2a   = (const float*)d_in[8];
    const float* W2b   = (const float*)d_in[9];
    const float* b2b   = (const float*)d_in[10];
    const float* Wc    = (const float*)d_in[11];
    const float* bc    = (const float*)d_in[12];
    const float* Wv    = (const float*)d_in[13];
    const float* bvv   = (const float*)d_in[14];
    const float* Wl    = (const float*)d_in[15];
    const float* bl    = (const float*)d_in[16];
    const float* Wz    = (const float*)d_in[17];
    const float* bz    = (const float*)d_in[18];

    float* out_x = (float*)d_out;                 // (N,H)
    float* out_Z = out_x + (size_t)NN * HH;       // (N,C,L)

    // workspace layout (floats; 32B-aligned regions)
    float* w = (float*)d_ws;
    float* deg   = w;               w += NN;               // deg -> dn (in place)
    float* ev    = w;               w += EE;
    float* Z1    = w;               w += (size_t)NN * LL;
    float* xh    = w;               w += (size_t)NN * HH;  // xh, later ln2
    float* bufB  = w;               w += (size_t)NN * HH;  // x1, later xh2
    float* bufC  = w;               w += (size_t)NN * HH;  // t,  later x1b
    float* cz    = w;               w += (size_t)NN * CC + 8;
    float* Zfull = w;               w += (size_t)NN * CC * LL;
    float* Z2    = w;               w += (size_t)NN * CC * LL;
    __bf16* Bp_emb = (__bf16*)w;                           // 16*4*512  = 32768
    __bf16* Bp_2a  = Bp_emb + 32768;                       // 16*8*512  = 65536 each
    __bf16* Bp_2b  = Bp_2a + 65536;
    __bf16* Bp_l   = Bp_2b + 65536;

    const int T = 256;
    const int gemm_grid = (NN + 63) / 64;     // 4 waves x 16 rows per block

    // --- pre-pack weights to bf16 fragment layout ---
    pack_b_kernel<<<(32768 + T - 1) / T, T, 0, stream>>>(W_emb, Bp_emb, INF_);
    pack_b_kernel<<<(65536 + T - 1) / T, T, 0, stream>>>(W2a,  Bp_2a, HH);
    pack_b_kernel<<<(65536 + T - 1) / T, T, 0, stream>>>(W2b,  Bp_2b, HH);
    pack_b_kernel<<<(65536 + T - 1) / T, T, 0, stream>>>(Wl,   Bp_l,  HH);

    // --- GCN normalization ---
    zero_kernel<<<(NN + T - 1) / T, T, 0, stream>>>(deg, NN);
    deg_kernel<<<(EE + T - 1) / T, T, 0, stream>>>(vals, row, deg);
    dn_kernel<<<(NN + T - 1) / T, T, 0, stream>>>(deg);
    ev_kernel<<<(EE + T - 1) / T, T, 0, stream>>>(vals, row, col, deg, ev);

    // --- noise thresholding ---
    nzt_kernel<<<NN, LL, 0, stream>>>(prob, Z1);

    // --- xh = x @ W_emb + b_emb ---
    wmma_gemm_kernel<INF_, 0><<<gemm_grid, 128, 0, stream>>>(x, Bp_emb, b_emb, nullptr, xh, NN);

    // --- x1 = LN(xh) ---
    layernorm_kernel<<<(NN + 3) / 4, 128, 0, stream>>>(xh, bufB);

    // --- t = silu(x1 @ W2a + b2a) ---
    wmma_gemm_kernel<HH, 1><<<gemm_grid, 128, 0, stream>>>(bufB, Bp_2a, b2a, nullptr, bufC, NN);

    // --- coeffZ = coeff*(x1 @ Wc + bc) ---
    coeffz_kernel<<<(NN + 3) / 4, 128, 0, stream>>>(bufB, Wc, bc, cz);

    // --- xh2 = t @ W2b + b2b + xh  (into bufB) ---
    wmma_gemm_kernel<HH, 2><<<gemm_grid, 128, 0, stream>>>(bufC, Bp_2b, b2b, xh, bufB, NN);

    // --- Zfull = Z1 + conv1x1(Wv,bv, silu(coeffZ*Z1)) ---
    zupdate_kernel<<<(NN * LL + T - 1) / T, T, 0, stream>>>(Z1, cz, Wv, bvv, Zfull);

    // --- ln2 = LN(xh2) (into xh) ; x1b = silu(ln2 @ Wl + bl) (into bufC) ---
    layernorm_kernel<<<(NN + 3) / 4, 128, 0, stream>>>(bufB, xh);
    wmma_gemm_kernel<HH, 1><<<gemm_grid, 128, 0, stream>>>(xh, Bp_l, bl, nullptr, bufC, NN);

    // --- out_x = xh2 + spmm(x1b) ---
    copy_kernel<<<((NN * HH) + T - 1) / T, T, 0, stream>>>(bufB, out_x, NN * HH);
    spmm_atomic_kernel<<<EE, HH, 0, stream>>>(bufC, row, col, ev, out_x, HH);

    // --- Z2 = silu(conv1x1(Wz,bz, vnorm(Zfull))) ; out_Z = Zfull + spmm(Z2) ---
    vnorm_kernel<<<(NN * LL + T - 1) / T, T, 0, stream>>>(Zfull, Wz, bz, Z2);
    copy_kernel<<<((NN * CC * LL) + T - 1) / T, T, 0, stream>>>(Zfull, out_Z, NN * CC * LL);
    spmm_atomic_kernel<<<EE, CC * LL, 0, stream>>>(Z2, row, col, ev, out_Z, CC * LL);
}
